// EnhancedWikiGraphSAGE_53730040873009
// MI455X (gfx1250) — matrix operand order repaired and checked by reading, not compiled, not run on hardware
//
#include <hip/hip_runtime.h>
#include <hip/hip_bf16.h>

typedef __attribute__((ext_vector_type(16))) __bf16 v16bf;
typedef __attribute__((ext_vector_type(8)))  __bf16 v8bf;
typedef __attribute__((ext_vector_type(8)))  float  v8f;

#define N_NODES 50000
#define E_EDGES 625000
#define IN_DIM  300
#define IN_PAD  320       // IN_DIM padded to multiple of 32
#define HID     128
#define NLAYERS 4
#define MT      5         // 16-row m-tiles per block (80 rows; 50000 = 625*80)
#define LDS_HW  136       // HID + 8 bf16 pad  (272 B/row -> bank-conflict-free)
#define LDS_XW  328       // IN_PAD + 8 bf16 pad (656 B/row)

// ---------------------------------------------------------------------------
// Fragment loaders (wave32 layouts per CDNA5 ISA §7.12.2)
// A 16x32 bf16: lane L -> row (L%16);
//   lanes 0-15:  elems 0..7 -> K=k0+0..7,  elems 8..15 -> K=k0+16..23
//   lanes 16-31: elems 0..7 -> K=k0+8..15, elems 8..15 -> K=k0+24..31
// ---------------------------------------------------------------------------

// A fragment from LDS (bf16, padded row stride): two ds_load_b128 per lane
__device__ __forceinline__ v16bf load_a_lds(const __bf16* __restrict__ sp,
                                            int strideElems, int k0, int lane) {
    int row = lane & 15;
    int hi  = (lane >> 4) & 1;
    const __bf16* rp = sp + row * strideElems;
    v8bf g0 = *(const v8bf*)(rp + k0 + hi * 8);
    v8bf g1 = *(const v8bf*)(rp + k0 + 16 + hi * 8);
    v16bf a;
#pragma unroll
    for (int e = 0; e < 8; ++e) { a[e] = g0[e]; a[8 + e] = g1[e]; }
    return a;
}

// B fragment from fragment-major packed buffer: one contiguous 32B per lane
__device__ __forceinline__ v16bf load_b_packed(const __bf16* __restrict__ p,
                                               int kt, int nt, int lane) {
    return *(const v16bf*)(p + ((size_t)((kt * 8 + nt) * 32 + lane)) * 16);
}

__device__ __forceinline__ v8bf cvt8(float4 a, float4 b, float s) {
    v8bf r;
    r[0] = (__bf16)(a.x * s); r[1] = (__bf16)(a.y * s);
    r[2] = (__bf16)(a.z * s); r[3] = (__bf16)(a.w * s);
    r[4] = (__bf16)(b.x * s); r[5] = (__bf16)(b.y * s);
    r[6] = (__bf16)(b.z * s); r[7] = (__bf16)(b.w * s);
    return r;
}

// ---------------------------------------------------------------------------
// Packing kernels (run once per launch; tiny)
// ---------------------------------------------------------------------------

// x [N, IN_DIM] fp32 -> xp [N, IN_PAD] bf16, zero-padded
__global__ void pack_x_kernel(const float* __restrict__ x, __bf16* __restrict__ xp) {
    long long i = (long long)blockIdx.x * blockDim.x + threadIdx.x;
    long long total = (long long)N_NODES * IN_PAD;
    long long stride = (long long)gridDim.x * blockDim.x;
    for (; i < total; i += stride) {
        int row = (int)(i / IN_PAD);
        int k   = (int)(i % IN_PAD);
        float f = (k < IN_DIM) ? x[(long long)row * IN_DIM + k] : 0.0f;
        xp[i] = (__bf16)f;
    }
}

// W [K, HID] fp32 row-major -> fragment-major bf16 [kt][nt=8][lane=32][e=16]
__global__ void pack_b_kernel(const float* __restrict__ W, int K, int Kt,
                              __bf16* __restrict__ out) {
    int t = blockIdx.x * blockDim.x + threadIdx.x;
    int total = Kt * 8 * 32;
    if (t >= total) return;
    int lane  = t & 31;
    int nt    = (t >> 5) & 7;
    int kt    = t >> 8;
    int col   = nt * 16 + (lane & 15);
    int kbase = kt * 32 + ((lane >> 4) & 1) * 8;
    __bf16* op = out + (size_t)t * 16;
#pragma unroll
    for (int e = 0; e < 16; ++e) {
        int k = kbase + (e < 8 ? e : e + 8);
        float f = (k < K) ? W[(size_t)k * HID + col] : 0.0f;
        op[e] = (__bf16)f;
    }
}

// ---------------------------------------------------------------------------
// Utility kernels
// ---------------------------------------------------------------------------
__global__ void zero_kernel(float* __restrict__ p, long long n) {
    long long i = (long long)blockIdx.x * blockDim.x + threadIdx.x;
    long long stride = (long long)gridDim.x * blockDim.x;
    for (; i < n; i += stride) p[i] = 0.0f;
}

__global__ void degree_kernel(const long long* __restrict__ dst,
                              float* __restrict__ deg) {
    long long e = (long long)blockIdx.x * blockDim.x + threadIdx.x;
    if (e >= E_EDGES) return;
    atomicAdd(&deg[dst[e]], 1.0f);
}

__global__ void invert_deg_kernel(float* __restrict__ deg) {
    int i = blockIdx.x * blockDim.x + threadIdx.x;
    if (i >= N_NODES) return;
    deg[i] = 1.0f / fmaxf(deg[i], 1.0f);
}

// ---------------------------------------------------------------------------
// Edge aggregation: one wave per edge, lane covers 4 features (32*4 = 128)
// ---------------------------------------------------------------------------
__global__ void __launch_bounds__(256)
sage_aggregate_kernel(const float* __restrict__ h,
                      const long long* __restrict__ src,
                      const long long* __restrict__ dst,
                      float* __restrict__ agg) {
    int lane = threadIdx.x & 31;
    long long e = (long long)blockIdx.x * (blockDim.x >> 5) + (threadIdx.x >> 5);
    if (e >= E_EDGES) return;
    long long s = src[e];
    long long d = dst[e];
    const float4* hp = (const float4*)(h + s * HID);
    float4 v = hp[lane];
    float* ap = agg + d * HID + lane * 4;
    atomicAdd(ap + 0, v.x);
    atomicAdd(ap + 1, v.y);
    atomicAdd(ap + 2, v.z);
    atomicAdd(ap + 3, v.w);
}

// ---------------------------------------------------------------------------
// Embedding GEMM: h = relu(xp @ Wp + b). 80 rows/block staged in LDS.
// Per kt: preload all 5 A fragments, then 5 independent WMMAs
// (lets the scheduler batch ds loads and use partial dscnt waits).
// ---------------------------------------------------------------------------
__global__ void __launch_bounds__(256)
sage_embed_kernel(const __bf16* __restrict__ xp, const __bf16* __restrict__ Wp,
                  const float* __restrict__ bias, float* __restrict__ h_out) {
    __shared__ __bf16 s_x[MT * 16 * LDS_XW];
    int t  = threadIdx.x;
    int m0 = blockIdx.x * (MT * 16);

    // Stage 80 rows x 320 bf16 from global -> LDS (16B chunks, coalesced)
    for (int c = t; c < MT * 16 * (IN_PAD / 8); c += 256) {
        int row = c / (IN_PAD / 8);
        int col = (c % (IN_PAD / 8)) * 8;
        v8bf v = *(const v8bf*)(xp + (size_t)(m0 + row) * IN_PAD + col);
        *(v8bf*)(s_x + row * LDS_XW + col) = v;
    }
    __syncthreads();

    int lane = t & 31;
    int nt   = t >> 5;
    v8f zero = {};
    v8f c[MT];
#pragma unroll
    for (int mt = 0; mt < MT; ++mt) c[mt] = zero;

    for (int kt = 0; kt < IN_PAD / 32; ++kt) {
        v16bf b = load_b_packed(Wp, kt, nt, lane);
        v16bf a[MT];
#pragma unroll
        for (int mt = 0; mt < MT; ++mt)
            a[mt] = load_a_lds(s_x + mt * 16 * LDS_XW, LDS_XW, kt * 32, lane);
#pragma unroll
        for (int mt = 0; mt < MT; ++mt)
            c[mt] = __builtin_amdgcn_wmma_f32_16x16x32_bf16(false, a[mt], false, b,
                                                            (short)0, c[mt], false, false);
    }

    int n  = nt * 16 + (lane & 15);
    float bv = bias[n];
    int hi = (lane >> 4) & 1;
#pragma unroll
    for (int mt = 0; mt < MT; ++mt) {
        int mb = m0 + mt * 16 + hi * 8;
#pragma unroll
        for (int r = 0; r < 8; ++r) {
            float v = c[mt][r] + bv;
            h_out[(long long)(mb + r) * HID + n] = v > 0.0f ? v : 0.0f;
        }
    }
}

// ---------------------------------------------------------------------------
// Fused SAGE layer: h_out = relu((inv_deg*agg) @ Wl + bl + h @ Wr)
// 80 rows/block staged in LDS as bf16 (inv_deg folded into agg tile).
// Depth-1 software pipeline: prefetch (mt+1) fragment pair during WMMA of mt,
// so each WMMA waits only on loads issued one step earlier (~4 frags in
// flight, keeps VGPRs ~130 alongside the 10 accumulator chains).
// ---------------------------------------------------------------------------
__global__ void __launch_bounds__(256)
sage_layer_kernel(const float* __restrict__ agg, const float* __restrict__ inv_deg,
                  const float* __restrict__ h,
                  const __bf16* __restrict__ Wlp, const float* __restrict__ bl,
                  const __bf16* __restrict__ Wrp,
                  float* __restrict__ h_out) {
    __shared__ __bf16 s_a[MT * 16 * LDS_HW];   // inv_deg * agg
    __shared__ __bf16 s_h[MT * 16 * LDS_HW];   // h
    int t  = threadIdx.x;
    int m0 = blockIdx.x * (MT * 16);

    // Stage: 80 rows x 128 cols, 8-elem chunks; 1280 chunks / 256 thr = 5 each
#pragma unroll
    for (int i = 0; i < MT; ++i) {
        int chunk = i * 256 + t;
        int row = chunk >> 4;            // 0..79
        int col = (chunk & 15) * 8;      // 0..120
        size_t goff = (size_t)(m0 + row) * HID + col;
        float sc = inv_deg[m0 + row];
        const float4* ap = (const float4*)(agg + goff);
        float4 a0 = ap[0], a1 = ap[1];
        *(v8bf*)(s_a + row * LDS_HW + col) = cvt8(a0, a1, sc);
        const float4* hp = (const float4*)(h + goff);
        float4 h0 = hp[0], h1 = hp[1];
        *(v8bf*)(s_h + row * LDS_HW + col) = cvt8(h0, h1, 1.0f);
    }
    __syncthreads();

    int lane = t & 31;
    int nt   = t >> 5;
    v8f zero = {};
    v8f c0[MT], c1[MT];
#pragma unroll
    for (int mt = 0; mt < MT; ++mt) { c0[mt] = zero; c1[mt] = zero; }

#pragma unroll
    for (int kt = 0; kt < HID / 32; ++kt) {
        v16bf b0 = load_b_packed(Wlp, kt, nt, lane);
        v16bf b1 = load_b_packed(Wrp, kt, nt, lane);
        v16bf a0 = load_a_lds(s_a, LDS_HW, kt * 32, lane);
        v16bf a1 = load_a_lds(s_h, LDS_HW, kt * 32, lane);
#pragma unroll
        for (int mt = 0; mt < MT; ++mt) {
            v16bf na0 = a0, na1 = a1;
            if (mt + 1 < MT) {
                na0 = load_a_lds(s_a + (mt + 1) * 16 * LDS_HW, LDS_HW, kt * 32, lane);
                na1 = load_a_lds(s_h + (mt + 1) * 16 * LDS_HW, LDS_HW, kt * 32, lane);
            }
            c0[mt] = __builtin_amdgcn_wmma_f32_16x16x32_bf16(false, a0, false, b0,
                                                             (short)0, c0[mt], false, false);
            c1[mt] = __builtin_amdgcn_wmma_f32_16x16x32_bf16(false, a1, false, b1,
                                                             (short)0, c1[mt], false, false);
            a0 = na0; a1 = na1;
        }
    }

    int n  = nt * 16 + (lane & 15);
    float bv = bl[n];
    int hi = (lane >> 4) & 1;
#pragma unroll
    for (int mt = 0; mt < MT; ++mt) {
        int mb = m0 + mt * 16 + hi * 8;
#pragma unroll
        for (int r = 0; r < 8; ++r) {
            float v = c0[mt][r] + c1[mt][r] + bv;
            h_out[(long long)(mb + r) * HID + n] = v > 0.0f ? v : 0.0f;
        }
    }
}

// ---------------------------------------------------------------------------
// Host-side launch orchestration
// ---------------------------------------------------------------------------
extern "C" void kernel_launch(void* const* d_in, const int* in_sizes, int n_in,
                              void* d_out, int out_size, void* d_ws, size_t ws_size,
                              hipStream_t stream) {
    const float*     x          = (const float*)d_in[0];
    const long long* edge_index = (const long long*)d_in[1];   // [2, E] int64
    const float*     emb_W      = (const float*)d_in[2];       // [300, 128]
    const float*     emb_b      = (const float*)d_in[3];
    const float*     Wl         = (const float*)d_in[4];       // [L, 128, 128]
    const float*     bl         = (const float*)d_in[5];       // [L, 128]
    const float*     Wr         = (const float*)d_in[6];       // [L, 128, 128]
    float*           out        = (float*)d_out;

    const long long* src = edge_index;
    const long long* dst = edge_index + E_EDGES;

    // Workspace layout (fp32 region first, then bf16 region)
    const size_t feat = (size_t)N_NODES * HID;
    float* h_a     = (float*)d_ws;
    float* h_b     = h_a + feat;
    float* agg     = h_b + feat;
    float* inv_deg = agg + feat;                       // N floats (deg accumulator)
    __bf16* xp     = (__bf16*)(inv_deg + N_NODES);     // [N, IN_PAD]
    __bf16* embWp  = xp + (size_t)N_NODES * IN_PAD;    // 10*8*32*16
    __bf16* wlp    = embWp + (size_t)(IN_PAD / 32) * 8 * 32 * 16;
    __bf16* wrp    = wlp + (size_t)NLAYERS * (HID / 32) * 8 * 32 * 16;

    const size_t wpack_elems = (size_t)(HID / 32) * 8 * 32 * 16;   // per matrix

    const int gemm_blocks = N_NODES / (MT * 16);  // 625
    const int edge_blocks = E_EDGES / 8;          // 78125 (8 waves/block)

    // --- degree -> inv_deg ---
    zero_kernel<<<512, 256, 0, stream>>>(inv_deg, (long long)N_NODES);
    degree_kernel<<<(E_EDGES + 255) / 256, 256, 0, stream>>>(dst, inv_deg);
    invert_deg_kernel<<<(N_NODES + 255) / 256, 256, 0, stream>>>(inv_deg);

    // --- pack inputs/weights to bf16 ---
    pack_x_kernel<<<4096, 256, 0, stream>>>(x, xp);
    {
        int tcnt = (IN_PAD / 32) * 8 * 32;
        pack_b_kernel<<<(tcnt + 255) / 256, 256, 0, stream>>>(emb_W, IN_DIM, IN_PAD / 32, embWp);
    }
    for (int l = 0; l < NLAYERS; ++l) {
        int tcnt = (HID / 32) * 8 * 32;
        pack_b_kernel<<<(tcnt + 255) / 256, 256, 0, stream>>>(
            Wl + (size_t)l * HID * HID, HID, HID / 32, wlp + (size_t)l * wpack_elems);
        pack_b_kernel<<<(tcnt + 255) / 256, 256, 0, stream>>>(
            Wr + (size_t)l * HID * HID, HID, HID / 32, wrp + (size_t)l * wpack_elems);
    }

    // --- embedding: h_a = relu(x @ emb_W + emb_b) ---
    sage_embed_kernel<<<gemm_blocks, 256, 0, stream>>>(xp, embWp, emb_b, h_a);

    // --- 4 SAGE layers with ping-pong buffers; last layer writes d_out ---
    float* hin  = h_a;
    float* hout = h_b;
    for (int l = 0; l < NLAYERS; ++l) {
        zero_kernel<<<2048, 256, 0, stream>>>(agg, (long long)feat);
        sage_aggregate_kernel<<<edge_blocks, 256, 0, stream>>>(hin, src, dst, agg);
        float* o = (l == NLAYERS - 1) ? out : hout;
        sage_layer_kernel<<<gemm_blocks, 256, 0, stream>>>(
            agg, inv_deg, hin,
            wlp + (size_t)l * wpack_elems, bl + (size_t)l * HID,
            wrp + (size_t)l * wpack_elems, o);
        float* tmp = hin; hin = hout; hout = tmp;
    }
}
